// KnnTransformer_18983755448332
// MI455X (gfx1250) — matrix operand order repaired and checked by reading, not compiled, not run on hardware
//
#include <hip/hip_runtime.h>
#include <hip/hip_bf16.h>
#include <math.h>

#define N_SRC 32768
#define N_Q   8192
#define DIM   128
#define HEADS 4
#define DH    64
#define INNER 256
#define MLPD  256
#define KNN   16

typedef __attribute__((ext_vector_type(16))) __bf16          bf16x16;
typedef __attribute__((ext_vector_type(8)))  float           f32x8;
typedef __attribute__((ext_vector_type(8)))  unsigned short  u16x8;
typedef __attribute__((ext_vector_type(16))) unsigned short  u16x16;

// ---------- scalar helpers ----------
__device__ __forceinline__ unsigned short f2bf(float f) {
    unsigned int u = __float_as_uint(f);
    u += 0x7fffu + ((u >> 16) & 1u);           // round to nearest even
    return (unsigned short)(u >> 16);
}
__device__ __forceinline__ float bf2f(unsigned short h) {
    return __uint_as_float(((unsigned int)h) << 16);
}
__device__ __forceinline__ float wave_sum(float v) {
    #pragma unroll
    for (int off = 16; off; off >>= 1) v += __shfl_xor(v, off, 32);
    return v;
}
__device__ __forceinline__ float gelu_tanh(float x) {
    float u = 0.7978845608028654f * (x + 0.044715f * x * x * x);
    float t = 2.f / (1.f + __expf(-2.f * u)) - 1.f;   // tanh(u)
    return 0.5f * x * (1.f + t);
}
__device__ __forceinline__ bf16x16 mk_frag(u16x8 lo, u16x8 hi) {
    u16x16 cat = __builtin_shufflevector(lo, hi, 0,1,2,3,4,5,6,7,8,9,10,11,12,13,14,15);
    return __builtin_bit_cast(bf16x16, cat);
}

// =====================================================================
// KNN: one wave per query. Per-lane sorted top-16 over 1024 points,
// then 16-round wave argmin merge through LDS.
// =====================================================================
__global__ void knn_kernel(const float* __restrict__ qpts,
                           const float* __restrict__ spts,
                           int* __restrict__ knn_idx) {
    __shared__ float lds_d[4][512];
    __shared__ int   lds_i[4][512];
    const int wave = threadIdx.x >> 5, lane = threadIdx.x & 31;
    const int qid  = blockIdx.x * 4 + wave;

    const float qx = qpts[qid * 3 + 0];
    const float qy = qpts[qid * 3 + 1];
    const float qz = qpts[qid * 3 + 2];

    float best[16]; int bidx[16];
    #pragma unroll
    for (int t = 0; t < 16; ++t) { best[t] = 3.4e38f; bidx[t] = 0; }

    for (int i = lane; i < N_SRC; i += 32) {
        float dx = spts[i * 3 + 0] - qx;
        float dy = spts[i * 3 + 1] - qy;
        float dz = spts[i * 3 + 2] - qz;
        float d  = dx * dx + dy * dy + dz * dz;
        if (d < best[15]) {
            best[15] = d; bidx[15] = i;
            #pragma unroll
            for (int t = 15; t > 0; --t) {
                if (best[t] < best[t - 1]) {
                    float td = best[t]; best[t] = best[t-1]; best[t-1] = td;
                    int   ti = bidx[t]; bidx[t] = bidx[t-1]; bidx[t-1] = ti;
                }
            }
        }
    }
    #pragma unroll
    for (int t = 0; t < 16; ++t) {
        lds_d[wave][lane * 16 + t] = best[t];
        lds_i[wave][lane * 16 + t] = bidx[t];
    }
    // merge: 16 rounds of wave argmin over per-lane sorted heads
    int p = 0;
    for (int r = 0; r < KNN; ++r) {
        float v = (p < 16) ? lds_d[wave][lane * 16 + p] : 3.4e38f;
        int winner = lane; float wv = v;
        #pragma unroll
        for (int off = 16; off; off >>= 1) {
            float ov = __shfl_xor(wv, off, 32);
            int   ol = __shfl_xor(winner, off, 32);
            if (ov < wv || (ov == wv && ol < winner)) { wv = ov; winner = ol; }
        }
        if (lane == winner) {
            knn_idx[qid * KNN + r] = lds_i[wave][lane * 16 + p];
            ++p;
        }
    }
}

// =====================================================================
// LayerNorm kernels (one wave per 128-wide row), bf16 output.
// Fourier pos: col j<64 -> sin(2pi xyz.B[:,j]); j>=64 -> cos(.., j-64)
// =====================================================================
__device__ __forceinline__ void ln_core(float v0, float v1, float v2, float v3,
                                        const float* g, const float* b,
                                        unsigned short* outr, int lane) {
    float s1 = wave_sum(v0 + v1 + v2 + v3);
    float s2 = wave_sum(v0*v0 + v1*v1 + v2*v2 + v3*v3);
    float mu = s1 * (1.f / 128.f);
    float var = s2 * (1.f / 128.f) - mu * mu;
    float inv = rsqrtf(var + 1e-5f);
    outr[lane +  0] = f2bf((v0 - mu) * inv * g[lane +  0] + b[lane +  0]);
    outr[lane + 32] = f2bf((v1 - mu) * inv * g[lane + 32] + b[lane + 32]);
    outr[lane + 64] = f2bf((v2 - mu) * inv * g[lane + 64] + b[lane + 64]);
    outr[lane + 96] = f2bf((v3 - mu) * inv * g[lane + 96] + b[lane + 96]);
}

__global__ void ln_xpos_kernel(const float* __restrict__ x,
                               const float* __restrict__ pts,
                               const float* __restrict__ B,
                               const float* __restrict__ g,
                               const float* __restrict__ b,
                               unsigned short* __restrict__ out) {
    const int wave = threadIdx.x >> 5, lane = threadIdx.x & 31;
    const int row  = blockIdx.x * 4 + wave;
    const float px = pts[row*3], py = pts[row*3+1], pz = pts[row*3+2];
    const float twopi = 6.283185307179586f;
    float pa = twopi * (px*B[lane]    + py*B[64+lane]    + pz*B[128+lane]);
    float pb = twopi * (px*B[lane+32] + py*B[64+lane+32] + pz*B[128+lane+32]);
    float sa, ca, sb, cb; __sincosf(pa, &sa, &ca); __sincosf(pb, &sb, &cb);
    const float* xr = x + (size_t)row * DIM;
    ln_core(xr[lane] + sa, xr[lane+32] + sb, xr[lane+64] + ca, xr[lane+96] + cb,
            g, b, out + (size_t)row * DIM, lane);
}

__global__ void ln_ypos_kernel(const float* __restrict__ sf,
                               const float* __restrict__ spts,
                               const int* __restrict__ knn_idx,
                               const float* __restrict__ B,
                               const float* __restrict__ g,
                               const float* __restrict__ b,
                               unsigned short* __restrict__ out) {
    const int wave = threadIdx.x >> 5, lane = threadIdx.x & 31;
    const int row  = blockIdx.x * 4 + wave;     // 0..131071
    const int idx  = knn_idx[row];
    const float px = spts[idx*3], py = spts[idx*3+1], pz = spts[idx*3+2];
    const float twopi = 6.283185307179586f;
    float pa = twopi * (px*B[lane]    + py*B[64+lane]    + pz*B[128+lane]);
    float pb = twopi * (px*B[lane+32] + py*B[64+lane+32] + pz*B[128+lane+32]);
    float sa, ca, sb, cb; __sincosf(pa, &sa, &ca); __sincosf(pb, &sb, &cb);
    const float* xr = sf + (size_t)idx * DIM;
    ln_core(xr[lane] + sa, xr[lane+32] + sb, xr[lane+64] + ca, xr[lane+96] + cb,
            g, b, out + (size_t)row * DIM, lane);
}

__global__ void ln_plain_kernel(const float* __restrict__ x,
                                const float* __restrict__ g,
                                const float* __restrict__ b,
                                unsigned short* __restrict__ out) {
    const int wave = threadIdx.x >> 5, lane = threadIdx.x & 31;
    const int row  = blockIdx.x * 4 + wave;
    const float* xr = x + (size_t)row * DIM;
    ln_core(xr[lane], xr[lane+32], xr[lane+64], xr[lane+96],
            g, b, out + (size_t)row * DIM, lane);
}

// =====================================================================
// bf16 WMMA GEMM: C[M,N] = act(A[M,K] * Wt[N,K]^T + bias) (+ residual)
// Block 256 thr = 8 waves; block tile 128x64; wave = one 16-row strip
// with 4 accumulators (16x64). A (128xK) and B (64xK) staged in dynamic
// LDS once (K <= 256), inner loop is pure ds_load + v_wmma.
// =====================================================================
#define GF_BIAS 1
#define GF_GELU 2
#define GF_RES  4
#define GF_F32  8
#define GF_BF16 16

__global__ void gemm_wmma_kernel(const unsigned short* __restrict__ A,
                                 const unsigned short* __restrict__ Wt,
                                 const float* __restrict__ bias,
                                 const float* __restrict__ residual,
                                 float* __restrict__ outF,
                                 unsigned short* __restrict__ outB,
                                 int M, int N, int K, int flags) {
    extern __shared__ __align__(16) unsigned short smem[];
    const int stride = K + 8;                    // rows stay 16B-aligned
    unsigned short* lds_a = smem;                // [128][stride]
    unsigned short* lds_b = smem + 128 * stride; // [64][stride]

    const int tid  = threadIdx.x;
    const int wave = tid >> 5, lane = tid & 31;
    const int m0 = blockIdx.x * 128;
    const int n0 = blockIdx.y * 64;
    const int kv8 = K >> 3;

    // stage A tile (128 x K)
    for (int c = tid; c < 128 * kv8; c += 256) {
        int r  = c / kv8;
        int kc = (c - r * kv8) << 3;
        *(u16x8*)&lds_a[r * stride + kc] = *(const u16x8*)(A + (size_t)(m0 + r) * K + kc);
    }
    // stage B tile (64 x K) = Wt rows n0..n0+63
    for (int c = tid; c < 64 * kv8; c += 256) {
        int r  = c / kv8;
        int kc = (c - r * kv8) << 3;
        *(u16x8*)&lds_b[r * stride + kc] = *(const u16x8*)(Wt + (size_t)(n0 + r) * K + kc);
    }
    __syncthreads();

    const int hi = lane >> 4;
    const int nl = lane & 15;
    const unsigned short* ar = lds_a + (wave * 16 + nl) * stride;

    f32x8 acc[4] = {};
    for (int kb = 0; kb < K; kb += 32) {
        const int abase = kb + hi * 8;           // A frag: K = abase+0..7, +16..23
        bf16x16 af = mk_frag(*(const u16x8*)(ar + abase),
                             *(const u16x8*)(ar + abase + 16));
        const int wbase = kb + hi * 8;           // B frag: K = wbase+0..7, +16..23
        #pragma unroll
        for (int nt = 0; nt < 4; ++nt) {
            const unsigned short* bp = lds_b + (nt * 16 + nl) * stride + wbase;
            bf16x16 wf = mk_frag(*(const u16x8*)bp, *(const u16x8*)(bp + 16));
            acc[nt] = __builtin_amdgcn_wmma_f32_16x16x32_bf16(false, af, false, wf,
                                                              (short)0, acc[nt],
                                                              false, false);
        }
    }

    #pragma unroll
    for (int nt = 0; nt < 4; ++nt) {
        const int nn = n0 + nt * 16 + nl;
        const float bb = (flags & GF_BIAS) ? bias[nn] : 0.f;
        #pragma unroll
        for (int j = 0; j < 8; ++j) {
            const int m = m0 + wave * 16 + j + hi * 8;
            float v = acc[nt][j] + bb;
            if (flags & GF_GELU) v = gelu_tanh(v);
            if (flags & GF_RES)  v += residual[(size_t)m * N + nn];
            if (flags & GF_F32)  outF[(size_t)m * N + nn] = v;
            if (flags & GF_BF16) outB[(size_t)m * N + nn] = f2bf(v);
        }
    }
}

// =====================================================================
// Attention: one wave per (query n, head h). K=16 neighbors, DH=64.
// q f32 [N_Q,256]; kv bf16 [N_Q*16,512] (k cols 0..255, v cols 256..511)
// =====================================================================
__global__ void attn_kernel(const float* __restrict__ q,
                            const unsigned short* __restrict__ kv,
                            unsigned short* __restrict__ out) {
    __shared__ float lds_a[8][16];
    const int wave = threadIdx.x >> 5, lane = threadIdx.x & 31;
    const int gw = blockIdx.x * 8 + wave;
    const int n = gw >> 2, h = gw & 3;

    const float* qp = q + (size_t)n * INNER + h * DH;
    const unsigned short* kvb = kv + (size_t)n * KNN * (2 * INNER);

    // dots: lanes 0..15 (and mirrored 16..31) each own one neighbor m
    const int m = lane & 15;
    const unsigned short* kp = kvb + (size_t)m * (2 * INNER) + h * DH;
    float s = 0.f;
    for (int d = 0; d < DH; d += 8) {
        u16x8 kk = *(const u16x8*)(kp + d);
        #pragma unroll
        for (int t = 0; t < 8; ++t) s += qp[d + t] * bf2f(kk[t]);
    }
    s *= 0.125f;                                    // DH^-0.5
    // softmax within each 16-lane half (xor offsets 8..1 stay in-half)
    float mx = s;
    #pragma unroll
    for (int off = 8; off; off >>= 1) mx = fmaxf(mx, __shfl_xor(mx, off, 32));
    float e = __expf(s - mx);
    float sm = e;
    #pragma unroll
    for (int off = 8; off; off >>= 1) sm += __shfl_xor(sm, off, 32);
    float a = e / sm;
    if (lane < 16) lds_a[wave][m] = a;              // same-wave LDS: in-order

    // out: lane covers d = lane and lane+32
    float o0 = 0.f, o1 = 0.f;
    for (int mm = 0; mm < KNN; ++mm) {
        float am = lds_a[wave][mm];
        const unsigned short* vp = kvb + (size_t)mm * (2 * INNER) + INNER + h * DH;
        o0 += am * bf2f(vp[lane]);
        o1 += am * bf2f(vp[lane + 32]);
    }
    out[(size_t)n * INNER + h * DH + lane]      = f2bf(o0);
    out[(size_t)n * INNER + h * DH + lane + 32] = f2bf(o1);
}

// =====================================================================
// Weight transpose + f32->bf16:  Wt[n*K+k] = bf16(W[k*N+n])
// =====================================================================
__global__ void prep_wt_kernel(const float* __restrict__ W,
                               unsigned short* __restrict__ Wt,
                               int Kdim, int N) {
    int i = blockIdx.x * 256 + threadIdx.x;
    if (i >= N * Kdim) return;
    int n = i / Kdim, k = i - n * Kdim;
    Wt[i] = f2bf(W[k * N + n]);
}

__global__ void copy_f32_kernel(const float* __restrict__ src,
                                float* __restrict__ dst, int n) {
    int i = blockIdx.x * 256 + threadIdx.x;
    if (i < n) dst[i] = src[i];
}

// =====================================================================
// Host orchestration
// =====================================================================
extern "C" void kernel_launch(void* const* d_in, const int* in_sizes, int n_in,
                              void* d_out, int out_size, void* d_ws, size_t ws_size,
                              hipStream_t stream) {
    (void)in_sizes; (void)n_in; (void)out_size; (void)ws_size;
    const float* src_feats    = (const float*)d_in[0];
    const float* src_points   = (const float*)d_in[1];
    const float* query_feats  = (const float*)d_in[2];
    const float* query_points = (const float*)d_in[3];
    const float* gauss_B      = (const float*)d_in[4];
    // per-layer pytree order (sorted keys):
    // W1, W2, Wkv, Wo, Wq, b1, b2, bo, ln_ca_b, ln_ca_g, ln_ff_b, ln_ff_g
    auto LP = [&](int layer, int slot) -> const float* {
        return (const float*)d_in[5 + layer * 12 + slot];
    };

    char* ws = (char*)d_ws;
    size_t off = 0;
    auto take = [&](size_t bytes) -> void* {
        void* p = ws + off;
        off += (bytes + 255) & ~(size_t)255;
        return p;
    };
    int*            knn_idx = (int*)           take((size_t)N_Q * KNN * 4);
    float*          x       = (float*)         take((size_t)N_Q * DIM * 4);
    unsigned short* xn      = (unsigned short*)take((size_t)N_Q * DIM * 2);
    unsigned short* yn      = (unsigned short*)take((size_t)N_Q * KNN * DIM * 2);
    float*          qbuf    = (float*)         take((size_t)N_Q * INNER * 4);
    unsigned short* kvbuf   = (unsigned short*)take((size_t)N_Q * KNN * 2 * INNER * 2);
    unsigned short* attn    = (unsigned short*)take((size_t)N_Q * INNER * 2);
    unsigned short* h1      = (unsigned short*)take((size_t)N_Q * MLPD * 2);
    unsigned short* wq_t    = (unsigned short*)take((size_t)INNER * DIM * 2);
    unsigned short* wkv_t   = (unsigned short*)take((size_t)2 * INNER * DIM * 2);
    unsigned short* wo_t    = (unsigned short*)take((size_t)DIM * INNER * 2);
    unsigned short* w1_t    = (unsigned short*)take((size_t)MLPD * DIM * 2);
    unsigned short* w2_t    = (unsigned short*)take((size_t)DIM * MLPD * 2);

    auto shmem = [](int K) -> size_t { return (size_t)192 * (K + 8) * 2; };

    // 1) KNN
    knn_kernel<<<N_Q / 4, 128, 0, stream>>>(query_points, src_points, knn_idx);
    // 2) x = query_feats
    copy_f32_kernel<<<(N_Q * DIM) / 256, 256, 0, stream>>>(query_feats, x, N_Q * DIM);

    for (int L = 0; L < 2; ++L) {
        const float* W1 = LP(L, 0);  const float* W2 = LP(L, 1);
        const float* Wkv = LP(L, 2); const float* Wo = LP(L, 3);
        const float* Wq = LP(L, 4);
        const float* b1 = LP(L, 5);  const float* b2 = LP(L, 6);
        const float* bo = LP(L, 7);
        const float* ln_ca_b = LP(L, 8);  const float* ln_ca_g = LP(L, 9);
        const float* ln_ff_b = LP(L, 10); const float* ln_ff_g = LP(L, 11);

        // weight prep (tiny)
        prep_wt_kernel<<<(INNER * DIM + 255) / 256, 256, 0, stream>>>(Wq,  wq_t,  DIM,   INNER);
        prep_wt_kernel<<<(2*INNER*DIM + 255) / 256, 256, 0, stream>>>(Wkv, wkv_t, DIM,   2*INNER);
        prep_wt_kernel<<<(DIM * INNER + 255) / 256, 256, 0, stream>>>(Wo,  wo_t,  INNER, DIM);
        prep_wt_kernel<<<(MLPD * DIM + 255) / 256, 256, 0, stream>>>(W1,  w1_t,  DIM,   MLPD);
        prep_wt_kernel<<<(DIM * MLPD + 255) / 256, 256, 0, stream>>>(W2,  w2_t,  MLPD,  DIM);

        // xn = LN(x + qpos)
        ln_xpos_kernel<<<N_Q / 4, 128, 0, stream>>>(x, query_points, gauss_B,
                                                    ln_ca_g, ln_ca_b, xn);
        // yn = LN(gather(src_feats) + npos)
        ln_ypos_kernel<<<(N_Q * KNN) / 4, 128, 0, stream>>>(src_feats, src_points, knn_idx,
                                                            gauss_B, ln_ca_g, ln_ca_b, yn);
        // q = xn @ Wq        [8192,128]x[128,256] -> f32
        gemm_wmma_kernel<<<dim3(N_Q / 128, INNER / 64), 256, shmem(DIM), stream>>>(
            xn, wq_t, nullptr, nullptr, qbuf, nullptr, N_Q, INNER, DIM, GF_F32);
        // kv = yn @ Wkv      [131072,128]x[128,512] -> bf16 (dominant GEMM)
        gemm_wmma_kernel<<<dim3((N_Q * KNN) / 128, (2 * INNER) / 64), 256, shmem(DIM), stream>>>(
            yn, wkv_t, nullptr, nullptr, nullptr, kvbuf, N_Q * KNN, 2 * INNER, DIM, GF_BF16);
        // attention -> attn (bf16)
        attn_kernel<<<(N_Q * HEADS) / 8, 256, 0, stream>>>(qbuf, kvbuf, attn);
        // x = attn @ Wo + bo + x
        gemm_wmma_kernel<<<dim3(N_Q / 128, DIM / 64), 256, shmem(INNER), stream>>>(
            attn, wo_t, bo, x, x, nullptr, N_Q, DIM, INNER, GF_BIAS | GF_RES | GF_F32);
        // xn = LN(x)  (ff norm)
        ln_plain_kernel<<<N_Q / 4, 128, 0, stream>>>(x, ln_ff_g, ln_ff_b, xn);
        // h1 = gelu(xn @ W1 + b1) -> bf16
        gemm_wmma_kernel<<<dim3(N_Q / 128, MLPD / 64), 256, shmem(DIM), stream>>>(
            xn, w1_t, b1, nullptr, nullptr, h1, N_Q, MLPD, DIM, GF_BIAS | GF_GELU | GF_BF16);
        // x = h1 @ W2 + b2 + x   (last layer writes d_out directly)
        float* xdst = (L == 1) ? (float*)d_out : x;
        gemm_wmma_kernel<<<dim3(N_Q / 128, DIM / 64), 256, shmem(MLPD), stream>>>(
            h1, w2_t, b2, x, xdst, nullptr, N_Q, DIM, MLPD, GF_BIAS | GF_RES | GF_F32);
    }
}
